// XRNNModel_11991548690508
// MI455X (gfx1250) — compile-verified, matrix operand order checked by reading, not run on hardware
//
#include <hip/hip_runtime.h>
#include <hip/hip_bf16.h>

// ---------------------------------------------------------------------------
// Problem constants (match the JAX reference)
// ---------------------------------------------------------------------------
namespace cfg {
constexpr int B    = 32;
constexpr int T    = 1024;
constexpr int NHID = 512;
constexpr int EMS  = 256;
constexpr int DIN  = 512;      // 2*EMS
constexpr int G3   = 3 * NHID; // 1536 (gate dim)
constexpr int V0   = 512;
constexpr int V1   = 512;
constexpr int NWG_REC = 16;    // persistent workgroups for recurrence

// LDS staging for the recurrence weights (per workgroup, 2 tile-sets)
constexpr int LDS_WAB_ELEMS = 2 * 16 * 1024;      // 64 KB as bf16
constexpr int LDS_WHH_ELEMS = 2 * 3 * 16 * NHID;  // 96 KB as bf16
constexpr size_t LDS_BYTES  = (size_t)(LDS_WAB_ELEMS + LDS_WHH_ELEMS) * 2;
}

typedef __bf16 bf16_t;
typedef __attribute__((ext_vector_type(16))) __bf16 v16bf;
typedef __attribute__((ext_vector_type(8)))  float  v8f;

// ---------------------------------------------------------------------------
// WMMA helpers (gfx1250, wave32). D = A(16x32 bf16) * B(32x16 bf16) + C(f32)
// A-frag per ISA 7.12.2: lane L holds row M=L&15; elements 0..7 = K(8*hi..),
// elements 8..15 = K(16+8*hi..), hi = lane>>4.
// B-frag: lane L holds weight row N=L&15 of row-major W[N][K] (x @ W^T form).
// C/D: element v -> row v + 8*hi, col lane&15.
// ---------------------------------------------------------------------------
__device__ __forceinline__ v8f wmma_bf16(v16bf a, v16bf b, v8f c) {
  return __builtin_amdgcn_wmma_f32_16x16x32_bf16(
      /*neg_a=*/false, a, /*neg_b=*/false, b,
      /*c_mod=*/(short)0, c, /*reuse_a=*/false, /*reuse_b=*/false);
}

// Load a 32-wide K slice of one matrix row (bf16, row-major) as a fragment.
// Works for global or LDS pointers (ds_load_b128 when shared-derived).
__device__ __forceinline__ v16bf frag_load_bf16(const bf16_t* p) {
  const int hi = (threadIdx.x >> 4) & 1;
  union { v16bf v; uint4 q[2]; } u;
  const bf16_t* q = p + 8 * hi;
  u.q[0] = *reinterpret_cast<const uint4*>(q);
  u.q[1] = *reinterpret_cast<const uint4*>(q + 16);
  return u.v;
}

__device__ __forceinline__ float sigmoidf_(float x) {
  return 1.0f / (1.0f + __expf(-x));
}

// Device-wide sense barrier (epoch counter). 16 resident WGs -> no deadlock.
__device__ __forceinline__ void grid_barrier(unsigned* cnt, unsigned* gen,
                                             unsigned nwg) {
  __syncthreads();
  if (threadIdx.x == 0) {
    __threadfence();
    unsigned g = __hip_atomic_load(gen, __ATOMIC_ACQUIRE, __HIP_MEMORY_SCOPE_AGENT);
    if (__hip_atomic_fetch_add(cnt, 1u, __ATOMIC_ACQ_REL, __HIP_MEMORY_SCOPE_AGENT)
        == nwg - 1) {
      __hip_atomic_store(cnt, 0u, __ATOMIC_RELAXED, __HIP_MEMORY_SCOPE_AGENT);
      __hip_atomic_fetch_add(gen, 1u, __ATOMIC_ACQ_REL, __HIP_MEMORY_SCOPE_AGENT);
    } else {
      while (__hip_atomic_load(gen, __ATOMIC_ACQUIRE, __HIP_MEMORY_SCOPE_AGENT) == g) {
        __builtin_amdgcn_s_sleep(2);
      }
    }
  }
  __syncthreads();
}

// ---------------------------------------------------------------------------
// Kernel 1: weight / table conversion + state init.
//   Wab[n][k] = bf16( k<512 ? A[n][k] : Bm[n][k-512] )   (512 x 1024)
//   Whh, Wih, decW0, decW1, emb0, emb1, default_h -> bf16
//   H[0] = h0 (f32 and bf16) ; barrier counters = 0
// ---------------------------------------------------------------------------
__global__ void prep_kernel(const float* __restrict__ Af,
                            const float* __restrict__ Bmf,
                            const float* __restrict__ Wihf,
                            const float* __restrict__ Whhf,
                            const float* __restrict__ dW0f,
                            const float* __restrict__ dW1f,
                            const float* __restrict__ emb0f,
                            const float* __restrict__ emb1f,
                            const float* __restrict__ defhf,
                            const float* __restrict__ h0,
                            bf16_t* __restrict__ Wab,
                            bf16_t* __restrict__ Whh,
                            bf16_t* __restrict__ Wih,
                            bf16_t* __restrict__ dW0,
                            bf16_t* __restrict__ dW1,
                            bf16_t* __restrict__ emb0b,
                            bf16_t* __restrict__ emb1b,
                            bf16_t* __restrict__ defhb,
                            float* __restrict__ Hf,
                            bf16_t* __restrict__ Hb,
                            unsigned* __restrict__ bar) {
  using namespace cfg;
  const size_t i = (size_t)blockIdx.x * blockDim.x + threadIdx.x;
  if (i == 0) { bar[0] = 0u; bar[1] = 0u; }
  if (i < (size_t)NHID * 1024) {
    const int n = (int)(i >> 10), k = (int)(i & 1023);
    const float v = (k < NHID) ? Af[n * NHID + k] : Bmf[n * NHID + (k - NHID)];
    Wab[i] = (bf16_t)v;
  }
  if (i < (size_t)G3 * NHID) Whh[i] = (bf16_t)Whhf[i];
  if (i < (size_t)G3 * DIN)  Wih[i] = (bf16_t)Wihf[i];
  if (i < (size_t)V0 * NHID) { dW0[i] = (bf16_t)dW0f[i]; dW1[i] = (bf16_t)dW1f[i]; }
  if (i < (size_t)V0 * EMS)  emb0b[i] = (bf16_t)emb0f[i];
  if (i < (size_t)V1 * EMS)  emb1b[i] = (bf16_t)emb1f[i];
  if (i < (size_t)NHID)      defhb[i] = (bf16_t)defhf[i];
  if (i < (size_t)B * NHID)  { Hf[i] = h0[i]; Hb[i] = (bf16_t)h0[i]; }
}

// ---------------------------------------------------------------------------
// Kernel 2: gi = embed(x) @ W_ih^T + b_ih, stored bf16, layout [T*B][1536]
// (row m = t*B + b). Embedding gather fused into the A-operand loader.
// Each wave: one 16-row M tile x 4 N tiles (A-frag reused across 4 WMMAs).
// ---------------------------------------------------------------------------
__global__ __launch_bounds__(256)
void gi_kernel(const int* __restrict__ tokens0,
               const int* __restrict__ tokens1,
               const bf16_t* __restrict__ emb0,
               const bf16_t* __restrict__ emb1,
               const bf16_t* __restrict__ Wih,
               const float* __restrict__ b_ih,
               bf16_t* __restrict__ gi) {
  using namespace cfg;
  const int lane = threadIdx.x & 31;
  const int r    = lane & 15;
  const int hi   = lane >> 4;
  const int grp  = blockIdx.x * 8 + (threadIdx.x >> 5);
  const int NGRP = G3 / 64;                 // 24 groups of 4 N-tiles
  const int ng    = grp % NGRP;
  const int mtile = grp / NGRP;
  const int m0 = mtile * 16, n0 = ng * 64;
  const int t = m0 >> 5;                    // B == 32: tile spans a single t
  const int b = (m0 & 31) + r;              // per-lane batch row
  const bf16_t* e0 = emb0 + (size_t)tokens0[b * T + t] * EMS;
  const bf16_t* e1 = emb1 + (size_t)tokens1[b * T + t] * EMS;
  const bf16_t* w0 = Wih + (size_t)(n0 + r) * DIN;

  v8f acc0 = {}, acc1 = {}, acc2 = {}, acc3 = {};
#pragma unroll 4
  for (int k0 = 0; k0 < DIN; k0 += 32) {
    const bf16_t* ar = (k0 < EMS) ? (e0 + k0) : (e1 + (k0 - EMS));
    v16bf a = frag_load_bf16(ar);
    acc0 = wmma_bf16(a, frag_load_bf16(w0 + k0), acc0);
    acc1 = wmma_bf16(a, frag_load_bf16(w0 + 16 * DIN + k0), acc1);
    acc2 = wmma_bf16(a, frag_load_bf16(w0 + 32 * DIN + k0), acc2);
    acc3 = wmma_bf16(a, frag_load_bf16(w0 + 48 * DIN + k0), acc3);
  }
#pragma unroll
  for (int j = 0; j < 4; ++j) {
    const v8f acc = (j == 0) ? acc0 : (j == 1) ? acc1 : (j == 2) ? acc2 : acc3;
    const int c = n0 + j * 16 + r;
    const float bias = b_ih[c];
#pragma unroll
    for (int v = 0; v < 8; ++v) {
      const int m = m0 + v + 8 * hi;
      gi[(size_t)m * G3 + c] = (bf16_t)(acc[v] + bias);
    }
  }
}

// ---------------------------------------------------------------------------
// Kernel 3: persistent recurrence. 16 WGs x 128 threads = 64 waves.
// Weight tiles are staged ONCE into LDS (160 KB/WG: 2 tile-sets of Wab and
// W_hh), then all 1024 steps run WMMA with ds_load_b128 operands.
// Phase 1: new_h = [h_last | prev] @ Wab^T (M=32,N=512,K=1024), prev gathered
//          per batch row from bf16 H[cond] (or default_h).
// Phase 2: gh = new_h_bf @ W_hh^T (3 gate tiles per wave, shared A-frag),
//          fuse gates with precomputed gi, write H[t+1] (f32 + bf16).
// 2 device barriers per step.
// ---------------------------------------------------------------------------
__global__ __launch_bounds__(128)
void recurrence_kernel(const int* __restrict__ conditions,
                       const int* __restrict__ skip_p,
                       const bf16_t* __restrict__ defhb,
                       const float* __restrict__ b_hh,
                       const bf16_t* __restrict__ Wab,
                       const bf16_t* __restrict__ Whh,
                       const bf16_t* __restrict__ gi,
                       float* __restrict__ Hf,
                       bf16_t* __restrict__ Hb,
                       float* __restrict__ newh_f,
                       bf16_t* __restrict__ newh_b,
                       unsigned* __restrict__ bar) {
  using namespace cfg;
  extern __shared__ char smem[];
  bf16_t* ldsWab = (bf16_t*)smem;                              // [2][16][1024]
  bf16_t* ldsWhh = (bf16_t*)(smem + (size_t)LDS_WAB_ELEMS * 2); // [2][3][16][512]

  const int lane = threadIdx.x & 31;
  const int r    = lane & 15;
  const int hi   = lane >> 4;
  const int lw   = threadIdx.x >> 5;                       // local wave 0..3
  const int wave = blockIdx.x * 4 + lw;                    // 0..63
  const int s    = lw >> 1;                                // tile-set 0/1
  const int skip = *skip_p;
  unsigned* cnt = bar;
  unsigned* gen = bar + 1;

  // ---- stage this WG's weight tiles into LDS (once) ----
  // Wab rows for set s: n = (blockIdx.x*2 + s)*16 + rr, k = 0..1023
  for (int idx = threadIdx.x; idx < 2 * 16 * (1024 / 8); idx += blockDim.x) {
    const int ss  = idx / (16 * 128);
    const int rem = idx % (16 * 128);
    const int rr  = rem / 128;
    const int kq  = rem % 128;
    const int n   = (blockIdx.x * 2 + ss) * 16 + rr;
    reinterpret_cast<uint4*>(ldsWab)[idx] =
        reinterpret_cast<const uint4*>(Wab + (size_t)n * 1024)[kq];
  }
  // Whh rows for set s, gate g: n = g*512 + (blockIdx.x*2 + s)*16 + rr
  for (int idx = threadIdx.x; idx < 2 * 3 * 16 * (NHID / 8); idx += blockDim.x) {
    const int ss   = idx / (3 * 16 * 64);
    const int rem  = idx % (3 * 16 * 64);
    const int g    = rem / (16 * 64);
    const int rem2 = rem % (16 * 64);
    const int rr   = rem2 / 64;
    const int kq   = rem2 % 64;
    const int n    = g * NHID + (blockIdx.x * 2 + ss) * 16 + rr;
    reinterpret_cast<uint4*>(ldsWhh)[idx] =
        reinterpret_cast<const uint4*>(Whh + (size_t)n * NHID)[kq];
  }
  __syncthreads();

  // per-wave LDS row pointers (lane r picks weight row within the tile)
  const bf16_t* wabRow = ldsWab + ((size_t)(s * 16 + r)) * 1024;
  const bf16_t* whrRow = ldsWhh + ((size_t)((s * 3 + 0) * 16 + r)) * NHID;
  const bf16_t* whzRow = ldsWhh + ((size_t)((s * 3 + 1) * 16 + r)) * NHID;
  const bf16_t* whnRow = ldsWhh + ((size_t)((s * 3 + 2) * 16 + r)) * NHID;

  const int m0 = (lw & 1) * 16;            // M tile (batch rows)
  const int n0 = (blockIdx.x * 2 + s) * 16; // N / column tile (same both phases)

  for (int t = 0; t < T; ++t) {
    // ---------------- phase 1: new_h ----------------
    {
      const int b = m0 + r;
      const bf16_t* hRow = Hb + ((size_t)t * B + b) * NHID;
      const int cond = conditions[b * T + t];
      const bool use = (t > skip) && (cond >= 0);
      const int cidx = cond > 0 ? cond : 0;
      const bf16_t* pRow = use ? (Hb + ((size_t)cidx * B + b) * NHID) : defhb;

      v8f acc = {};
#pragma unroll 4
      for (int k0 = 0; k0 < 1024; k0 += 32) {
        v16bf a = (k0 < NHID) ? frag_load_bf16(hRow + k0)
                              : frag_load_bf16(pRow + (k0 - NHID));
        acc = wmma_bf16(a, frag_load_bf16(wabRow + k0), acc);
      }
      const int c = n0 + r;
#pragma unroll
      for (int v = 0; v < 8; ++v) {
        const int row = m0 + v + 8 * hi;      // batch index
        newh_f[row * NHID + c] = acc[v];
        newh_b[row * NHID + c] = (bf16_t)acc[v];
      }
    }
    grid_barrier(cnt, gen, NWG_REC);

    // ---------------- phase 2: GRU gates ----------------
    {
      const int b = m0 + r;
      const bf16_t* aRow = newh_b + (size_t)b * NHID;

      v8f accr = {}, accz = {}, accn = {};
#pragma unroll 2
      for (int k0 = 0; k0 < NHID; k0 += 32) {
        v16bf a = frag_load_bf16(aRow + k0);  // shared across 3 gate tiles
        accr = wmma_bf16(a, frag_load_bf16(whrRow + k0), accr);
        accz = wmma_bf16(a, frag_load_bf16(whzRow + k0), accz);
        accn = wmma_bf16(a, frag_load_bf16(whnRow + k0), accn);
      }
      const int c = n0 + r;
      const float bhr = b_hh[c], bhz = b_hh[NHID + c], bhn = b_hh[2 * NHID + c];
#pragma unroll
      for (int v = 0; v < 8; ++v) {
        const int row = m0 + v + 8 * hi;      // batch index
        const bf16_t* gir = gi + ((size_t)t * B + row) * G3;
        const float ir  = (float)gir[c];
        const float iz  = (float)gir[NHID + c];
        const float inn = (float)gir[2 * NHID + c];
        const float rg = sigmoidf_(ir + accr[v] + bhr);
        const float zg = sigmoidf_(iz + accz[v] + bhz);
        const float ng = tanhf(inn + rg * (accn[v] + bhn));
        const float nh = newh_f[row * NHID + c];
        const float h = (1.0f - zg) * ng + zg * nh;
        Hf[((size_t)(t + 1) * B + row) * NHID + c] = h;
        Hb[((size_t)(t + 1) * B + row) * NHID + c] = (bf16_t)h;
      }
    }
    grid_barrier(cnt, gen, NWG_REC);
  }
}

// ---------------------------------------------------------------------------
// Kernel 4: decoders. dec[m][v] = H[t+1][b] @ decW^T + decb, m = b*T + t.
// A operand from bf16 H; each wave: 1 M tile x 4 N tiles.
// ---------------------------------------------------------------------------
__global__ __launch_bounds__(256)
void dec_kernel(const bf16_t* __restrict__ Hb,
                const bf16_t* __restrict__ W,
                const float* __restrict__ bias,
                float* __restrict__ out) {
  using namespace cfg;
  const int lane = threadIdx.x & 31;
  const int r    = lane & 15;
  const int hi   = lane >> 4;
  const int grp  = blockIdx.x * 8 + (threadIdx.x >> 5);
  const int NGRP = V0 / 64;                // 8 groups of 4 N-tiles
  const int ng    = grp % NGRP;
  const int mtile = grp / NGRP;
  const int m0 = mtile * 16, n0 = ng * 64;
  const int bb = m0 >> 10;                 // T == 1024
  const int t  = (m0 & 1023) + r;          // per-lane time row
  const bf16_t* aRow = Hb + ((size_t)(t + 1) * B + bb) * NHID;
  const bf16_t* w0 = W + (size_t)(n0 + r) * NHID;

  v8f acc0 = {}, acc1 = {}, acc2 = {}, acc3 = {};
#pragma unroll 4
  for (int k0 = 0; k0 < NHID; k0 += 32) {
    v16bf a = frag_load_bf16(aRow + k0);
    acc0 = wmma_bf16(a, frag_load_bf16(w0 + k0), acc0);
    acc1 = wmma_bf16(a, frag_load_bf16(w0 + 16 * NHID + k0), acc1);
    acc2 = wmma_bf16(a, frag_load_bf16(w0 + 32 * NHID + k0), acc2);
    acc3 = wmma_bf16(a, frag_load_bf16(w0 + 48 * NHID + k0), acc3);
  }
#pragma unroll
  for (int j = 0; j < 4; ++j) {
    const v8f acc = (j == 0) ? acc0 : (j == 1) ? acc1 : (j == 2) ? acc2 : acc3;
    const int c = n0 + j * 16 + r;
    const float bv = bias[c];
#pragma unroll
    for (int v = 0; v < 8; ++v) {
      const int m = m0 + v + 8 * hi;
      out[(size_t)m * V0 + c] = acc[v] + bv;
    }
  }
}

// Kernel 5: out[:, -1] = H[T]  (fp32 state)
__global__ void last_kernel(const float* __restrict__ Hf, float* __restrict__ out) {
  using namespace cfg;
  const int i = blockIdx.x * blockDim.x + threadIdx.x;
  if (i < B * NHID) out[i] = Hf[(size_t)T * B * NHID + i];
}

// ---------------------------------------------------------------------------
// Host launcher
// ---------------------------------------------------------------------------
extern "C" void kernel_launch(void* const* d_in, const int* in_sizes, int n_in,
                              void* d_out, int out_size, void* d_ws, size_t ws_size,
                              hipStream_t stream) {
  using namespace cfg;
  const int*   tokens0    = (const int*)d_in[0];
  const int*   tokens1    = (const int*)d_in[1];
  const int*   conditions = (const int*)d_in[2];
  const float* h0         = (const float*)d_in[3];
  const float* emb0       = (const float*)d_in[4];
  const float* emb1       = (const float*)d_in[5];
  const float* Af         = (const float*)d_in[6];
  const float* Bmf        = (const float*)d_in[7];
  const float* default_h  = (const float*)d_in[8];
  const float* W_ih       = (const float*)d_in[9];
  const float* W_hh       = (const float*)d_in[10];
  const float* b_ih       = (const float*)d_in[11];
  const float* b_hh       = (const float*)d_in[12];
  const float* decW0      = (const float*)d_in[13];
  const float* decb0      = (const float*)d_in[14];
  const float* decW1      = (const float*)d_in[15];
  const float* decb1      = (const float*)d_in[16];
  const int*   skip       = (const int*)d_in[17];
  (void)in_sizes; (void)n_in; (void)out_size; (void)ws_size;

  char* ws = (char*)d_ws;
  size_t off = 0;
  auto take = [&](size_t bytes) -> char* {
    char* p = ws + off;
    off = (off + bytes + 255) & ~(size_t)255;
    return p;
  };
  unsigned* bar    = (unsigned*)take(256);
  float*    Hf     = (float*)  take(sizeof(float)  * (size_t)(T + 1) * B * NHID);
  bf16_t*   Hb     = (bf16_t*) take(sizeof(bf16_t) * (size_t)(T + 1) * B * NHID);
  bf16_t*   gi     = (bf16_t*) take(sizeof(bf16_t) * (size_t)T * B * G3);
  bf16_t*   Wab    = (bf16_t*) take(sizeof(bf16_t) * (size_t)NHID * 1024);
  bf16_t*   Whh    = (bf16_t*) take(sizeof(bf16_t) * (size_t)G3 * NHID);
  bf16_t*   Wih    = (bf16_t*) take(sizeof(bf16_t) * (size_t)G3 * DIN);
  bf16_t*   dW0    = (bf16_t*) take(sizeof(bf16_t) * (size_t)V0 * NHID);
  bf16_t*   dW1    = (bf16_t*) take(sizeof(bf16_t) * (size_t)V1 * NHID);
  bf16_t*   emb0b  = (bf16_t*) take(sizeof(bf16_t) * (size_t)V0 * EMS);
  bf16_t*   emb1b  = (bf16_t*) take(sizeof(bf16_t) * (size_t)V1 * EMS);
  bf16_t*   defhb  = (bf16_t*) take(sizeof(bf16_t) * (size_t)NHID);
  float*    newh_f = (float*)  take(sizeof(float)  * (size_t)B * NHID);
  bf16_t*   newh_b = (bf16_t*) take(sizeof(bf16_t) * (size_t)B * NHID);

  // 1) pack/convert weights + tables, init H[0] and barrier
  {
    const int total = G3 * DIN;  // largest conversion task
    prep_kernel<<<(total + 255) / 256, 256, 0, stream>>>(
        Af, Bmf, W_ih, W_hh, decW0, decW1, emb0, emb1, default_h, h0,
        Wab, Whh, Wih, dW0, dW1, emb0b, emb1b, defhb, Hf, Hb, bar);
  }
  // 2) gi = embed(x) @ W_ih^T + b_ih   (M=32768, N=1536, K=512)
  {
    const int grps = (T * B / 16) * (G3 / 64);  // 2048 * 24
    gi_kernel<<<grps / 8, 256, 0, stream>>>(tokens0, tokens1, emb0b, emb1b,
                                            Wih, b_ih, gi);
  }
  // 3) persistent sequential recurrence (1024 steps, device barrier,
  //    weights resident in 160 KB LDS per workgroup)
  recurrence_kernel<<<NWG_REC, 128, LDS_BYTES, stream>>>(
      conditions, skip, defhb, b_hh, Wab, Whh, gi, Hf, Hb, newh_f, newh_b, bar);

  // 4) decoders + 5) final hidden state
  float* out = (float*)d_out;
  {
    const int grps = (T * B / 16) * (V0 / 64);  // 2048 * 8
    dec_kernel<<<grps / 8, 256, 0, stream>>>(Hb, dW0, decb0, out);
    dec_kernel<<<grps / 8, 256, 0, stream>>>(Hb, dW1, decb1,
                                             out + (size_t)B * T * V0);
  }
  last_kernel<<<(B * NHID + 255) / 256, 256, 0, stream>>>(
      Hf, out + (size_t)2 * B * T * V0);
}
